// LatexOCRModel_90975997264330
// MI455X (gfx1250) — compile-verified, matrix operand order checked by reading, not run on hardware
//
#include <hip/hip_runtime.h>
#include <hip/hip_bf16.h>

// MI455X / gfx1250, wave32. WMMA f32<=f16 16x16x32, 4-way N-blocked,
// NHWC activations (contiguous-channel b128 loads), branchless inner loops.

typedef __attribute__((ext_vector_type(16))) _Float16 v16h;
typedef __attribute__((ext_vector_type(8)))  _Float16 h8;
typedef __attribute__((ext_vector_type(8)))  float    v8f;
typedef __attribute__((ext_vector_type(4)))  float    f4v;

#define EPSV 1e-5f

__device__ __forceinline__ float sigmoidf_(float x) { return 1.0f / (1.0f + __expf(-x)); }

// WMMA f16 operand element i -> K offset within a 32-wide chunk (ISA 7.12.2):
// lanes 0-15 hold K = {0..7, 16..23}, lanes 16-31 hold K = {8..15, 24..31}.
__device__ __forceinline__ int koff(int i, int half) {
  return ((i < 8) ? i : i + 8) + (half << 3);
}

// ---------------------------------------------------------------------------
// Layout / weight preparation (once per call; deterministic)
// ---------------------------------------------------------------------------
__global__ void nchw_to_nhwc(const float* __restrict__ src, float* __restrict__ dst,
                             int C, int HW, int total)
{
  const int idx = blockIdx.x * blockDim.x + threadIdx.x;
  if (idx >= total) return;
  const int c = idx % C;
  int t = idx / C;
  const int p = t % HW;
  const int b = t / HW;
  dst[idx] = src[((size_t)b * C + c) * HW + p];
}

__global__ void cvt_f16(const float* __restrict__ src, _Float16* __restrict__ dst, int n)
{
  const int idx = blockIdx.x * blockDim.x + threadIdx.x;
  if (idx < n) dst[idx] = (_Float16)src[idx];
}

// OIHW fp32 -> [KHW][OC][IC] f16 (ic contiguous per (kpos, oc) row)
__global__ void repack_convw(const float* __restrict__ w, _Float16* __restrict__ out,
                             int OC, int IC, int KHW, int total)
{
  const int idx = blockIdx.x * blockDim.x + threadIdx.x;
  if (idx >= total) return;
  const int kpos = idx % KHW;
  const int t    = idx / KHW;
  const int ic   = t % IC;
  const int oc   = t / IC;
  out[((size_t)kpos * OC + oc) * IC + ic] = (_Float16)w[idx];
}

// ---------------------------------------------------------------------------
// Implicit-GEMM conv + BN (+res) (+ReLU). NHWC in / NHWC out.
// One wave -> 16(M=pixels) x 64(N=OC) block. Call-site invariants:
// OC % 64 == 0, OW % 16 == 0, (OH*OW) % 16 == 0, M % 16 == 0.
// (ky,kx) outer / ic chunks inner; channels contiguous -> b128 A loads.
// ---------------------------------------------------------------------------
__global__ __launch_bounds__(32) void conv_bn_wmma(
    const float* __restrict__ x, const _Float16* __restrict__ wp,
    const float* __restrict__ bng, const float* __restrict__ bnb,
    const float* __restrict__ bnm, const float* __restrict__ bnv,
    const float* __restrict__ res, float* __restrict__ out,
    int IC, int IH, int IW, int OC, int OH, int OW,
    int KH, int KW, int stride, int pad, int relu)
{
  const int lane = threadIdx.x & 31;
  const int half = lane >> 4;
  const int l15  = lane & 15;
  const int nTiles = OC >> 6;
  const int mT = (int)blockIdx.x / nTiles;
  const int nT = (int)blockIdx.x % nTiles;
  const int mBase = mT << 4, nBase = nT << 6;

  // Tile-uniform pixel decomposition (OW, OH*OW multiples of 16):
  const int hw  = OH * OW;
  const int b_  = mBase / hw;
  const int rm  = mBase % hw;
  const int oy  = rm / OW;          // uniform across the tile
  const int oxb = rm % OW;          // multiple of 16
  const int ox  = oxb + l15;        // per-lane A-fragment pixel column
  const int kb  = half << 3;

  const float* xb0 = x + (size_t)b_ * IH * IW * IC;   // + ((iy*IW+ix)*IC + ic)

  v8f acc[4] = {{}, {}, {}, {}};

  for (int ky = 0; ky < KH; ++ky) {
    const int iy = oy * stride - pad + ky;
    if (iy < 0 || iy >= IH) continue;              // uniform skip: zero taps
    for (int kx = 0; kx < KW; ++kx) {
      const int ix  = ox * stride - pad + kx;
      const int ixc = (ix < 0) ? 0 : ((ix >= IW) ? IW - 1 : ix);
      const float am = (ix >= 0 && ix < IW) ? 1.0f : 0.0f;
      const float*    xb = xb0 + ((size_t)iy * IW + ixc) * IC;
      const _Float16* wk = wp + (size_t)(ky * KW + kx) * OC * IC;

      for (int ic0 = 0; ic0 < IC; ic0 += 32) {
        v16h a, bm[4];
        if (ic0 + 32 <= IC) {                      // fast path (IC % 32 == 0)
          // A: channels contiguous -> two 32B runs -> 4x b128 loads
          const f4v x0 = *(const f4v*)(xb + ic0 + kb);
          const f4v x1 = *(const f4v*)(xb + ic0 + kb + 4);
          const f4v x2 = *(const f4v*)(xb + ic0 + kb + 16);
          const f4v x3 = *(const f4v*)(xb + ic0 + kb + 20);
#pragma unroll
          for (int i = 0; i < 4; ++i) {
            a[i]      = (_Float16)(x0[i] * am);
            a[i + 4]  = (_Float16)(x1[i] * am);
            a[i + 8]  = (_Float16)(x2[i] * am);
            a[i + 12] = (_Float16)(x3[i] * am);
          }
#pragma unroll
          for (int j = 0; j < 4; ++j) {
            const _Float16* wr = wk + (nBase + j * 16 + l15) * IC + ic0 + kb;
            const h8 lo = *(const h8*)(wr);        // 16B aligned
            const h8 hi = *(const h8*)(wr + 16);
#pragma unroll
            for (int i = 0; i < 8; ++i) { bm[j][i] = lo[i]; bm[j][i + 8] = hi[i]; }
          }
        } else {                                   // tail (conv1, IC=3)
#pragma unroll
          for (int i = 0; i < 16; ++i) {
            const int ic  = ic0 + koff(i, half);
            const int icc = (ic < IC) ? ic : IC - 1;
            const float mz = (ic < IC) ? am : 0.0f;
            a[i] = (_Float16)(xb[icc] * mz);       // garbage B x 0 == 0
          }
#pragma unroll
          for (int j = 0; j < 4; ++j) {
            const _Float16* wr = wk + (nBase + j * 16 + l15) * IC;
#pragma unroll
            for (int i = 0; i < 16; ++i) {
              const int ic  = ic0 + koff(i, half);
              const int icc = (ic < IC) ? ic : IC - 1;
              bm[j][i] = wr[icc];                  // unconditional load
            }
          }
        }
        // single WMMA call site -> no accumulator register rotation
#pragma unroll
        for (int j = 0; j < 4; ++j)
          acc[j] = __builtin_amdgcn_wmma_f32_16x16x32_f16(
              false, a, false, bm[j], (short)0, acc[j], false, false);
      }
    }
  }

  // epilogue: BN, optional residual, ReLU. Rows share (b_, oy); ox varies.
  float scl[4], mu[4], bt[4];
#pragma unroll
  for (int j = 0; j < 4; ++j) {
    const int oc = nBase + j * 16 + l15;
    scl[j] = bng[oc] * rsqrtf(bnv[oc] + EPSV);
    mu[j]  = bnm[oc];
    bt[j]  = bnb[oc];
  }
#pragma unroll
  for (int r = 0; r < 8; ++r) {
    const int ox2 = oxb + r + (half << 3);         // < OW (OW mult of 16)
    const size_t pix = (size_t)(b_ * hw + oy * OW + ox2) * OC;
#pragma unroll
    for (int j = 0; j < 4; ++j) {
      const int oc = nBase + j * 16 + l15;
      float v = (acc[j][r] - mu[j]) * scl[j] + bt[j];
      if (res) v += res[pix + oc];                 // NHWC: lane-coalesced
      if (relu) v = fmaxf(v, 0.0f);
      out[pix + oc] = v;
    }
  }
}

// ---------------------------------------------------------------------------
// Dense GEMM: C[M,N] = (accum? C:0) + A[M,K](f32) * Wh[N,K](f16)^T + bias[N]
// One wave -> 16 x 64 block. Requirements: M % 16 == 0, K % 32 == 0.
// N-tail rows read a clamped valid row; their accumulators are never stored.
// ---------------------------------------------------------------------------
__global__ __launch_bounds__(32) void gemm_wmma(
    const float* __restrict__ A, int lda,
    const _Float16* __restrict__ Wh,
    const float* __restrict__ bias,
    float* __restrict__ C, int ldc,
    int N, int K, int relu, int accum)
{
  const int lane = threadIdx.x & 31;
  const int half = lane >> 4, l15 = lane & 15;
  const int nTiles = (N + 63) >> 6;
  const int mT = (int)blockIdx.x / nTiles, nT = (int)blockIdx.x % nTiles;
  const int mBase = mT << 4, nBase = nT << 6;
  const int kb = half << 3;
  const float* Ar = A + (size_t)(mBase + l15) * lda;

  int nrow[4]; bool nok[4]; const _Float16* wr[4];
#pragma unroll
  for (int j = 0; j < 4; ++j) {
    nrow[j] = nBase + j * 16 + l15;
    nok[j]  = nrow[j] < N;
    wr[j]   = Wh + (size_t)(nok[j] ? nrow[j] : N - 1) * K;   // clamped, valid
  }

  v8f acc[4] = {{}, {}, {}, {}};
  for (int kk = 0; kk < K; kk += 32) {
    if (kk + 256 < K) __builtin_prefetch(wr[0] + kk + 256, 0, 1);
    const f4v a0 = *(const f4v*)(Ar + kk + kb);
    const f4v a1 = *(const f4v*)(Ar + kk + kb + 4);
    const f4v a2 = *(const f4v*)(Ar + kk + kb + 16);
    const f4v a3 = *(const f4v*)(Ar + kk + kb + 20);
    v16h a;
#pragma unroll
    for (int i = 0; i < 4; ++i) {
      a[i]      = (_Float16)a0[i];
      a[i + 4]  = (_Float16)a1[i];
      a[i + 8]  = (_Float16)a2[i];
      a[i + 12] = (_Float16)a3[i];
    }
    v16h bm[4];
#pragma unroll
    for (int j = 0; j < 4; ++j) {
      const _Float16* p = wr[j] + kk + kb;       // 16B aligned
      const h8 lo = *(const h8*)(p);
      const h8 hi = *(const h8*)(p + 16);
#pragma unroll
      for (int i = 0; i < 8; ++i) { bm[j][i] = lo[i]; bm[j][i + 8] = hi[i]; }
    }
#pragma unroll
    for (int j = 0; j < 4; ++j)
      acc[j] = __builtin_amdgcn_wmma_f32_16x16x32_f16(
          false, a, false, bm[j], (short)0, acc[j], false, false);
  }

#pragma unroll
  for (int j = 0; j < 4; ++j) {
    if (!nok[j]) continue;
    const int n = nrow[j];
    const float bz = bias ? bias[n] : 0.0f;
#pragma unroll
    for (int r = 0; r < 8; ++r) {
      const int mo = mBase + r + (half << 3);    // < M (M mult of 16)
      float v = acc[j][r] + bz;
      if (accum) v += C[(size_t)mo * ldc + n];
      if (relu)  v = fmaxf(v, 0.0f);
      C[(size_t)mo * ldc + n] = v;
    }
  }
}

// ---------------------------------------------------------------------------
// 3x3 stride-2 pad-1 max pool, NHWC (channel contiguous -> coalesced)
// ---------------------------------------------------------------------------
__global__ void maxpool3x3s2_nhwc(const float* __restrict__ x, float* __restrict__ y,
                                  int C, int IH, int IW, int OH, int OW, int total)
{
  const int idx = blockIdx.x * blockDim.x + threadIdx.x;
  if (idx >= total) return;
  const int c = idx % C;
  int t = idx / C;
  const int ox = t % OW; t /= OW;
  const int oy = t % OH; const int b = t / OH;
  float mv = -3.4e38f;
#pragma unroll
  for (int ky = 0; ky < 3; ++ky) {
#pragma unroll
    for (int kx = 0; kx < 3; ++kx) {
      const int iy = oy * 2 - 1 + ky;
      const int ix = ox * 2 - 1 + kx;
      if (iy >= 0 && iy < IH && ix >= 0 && ix < IW)
        mv = fmaxf(mv, x[((size_t)(b * IH + iy) * IW + ix) * C + c]);
    }
  }
  y[idx] = mv;
}

// ---------------------------------------------------------------------------
// Embedding gather: embA[b,t,:] = emb[captions[b,t]]
// ---------------------------------------------------------------------------
__global__ void embed_gather(const int* __restrict__ captions,
                             const float* __restrict__ emb,
                             float* __restrict__ out)
{
  const int idx = blockIdx.x * blockDim.x + threadIdx.x;
  const int total = 16 * 128 * 256;
  if (idx >= total) return;
  const int e  = idx % 256;
  const int bt = idx / 256;
  const int tok = captions[bt];
  out[idx] = emb[(size_t)tok * 256 + e];
}

// ---------------------------------------------------------------------------
// Fused attention step for one t (B=16, P=64, ATT=ENC=EMB=256):
//   e[p] = relu(attE[b,p,:]+hproj[b,:]) . full_w + full_b
//   alpha = softmax_p(e); ctx[d] = sum_p alpha[p]*feats[b,p,d]
//   xin[b] = concat(embA[b,t,:], ctx)
// ---------------------------------------------------------------------------
__global__ __launch_bounds__(256) void attention_ctx(
    const float* __restrict__ attE, const float* __restrict__ hproj,
    const float* __restrict__ feats, const float* __restrict__ fullw,
    const float* __restrict__ fullb, const float* __restrict__ embA,
    int t, float* __restrict__ xin)
{
  const int b = blockIdx.x, tid = threadIdx.x;
  __shared__ float e[64];
  __shared__ float alpha[64];
  if (tid < 64) {
    const float* ae = attE  + ((size_t)b * 64 + tid) * 256;
    const float* hp = hproj + (size_t)b * 256;
    float s = 0.0f;
    for (int a_ = 0; a_ < 256; ++a_) {
      float v = ae[a_] + hp[a_];
      v = fmaxf(v, 0.0f);
      s += v * fullw[a_];
    }
    e[tid] = s + fullb[0];
  }
  __syncthreads();
  if (tid == 0) {
    float mx = e[0];
    for (int p = 1; p < 64; ++p) mx = fmaxf(mx, e[p]);
    float sum = 0.0f;
    for (int p = 0; p < 64; ++p) { alpha[p] = __expf(e[p] - mx); sum += alpha[p]; }
    const float inv = 1.0f / sum;
    for (int p = 0; p < 64; ++p) alpha[p] *= inv;
  }
  __syncthreads();
  float s = 0.0f;
  for (int p = 0; p < 64; ++p)
    s += alpha[p] * feats[((size_t)b * 64 + p) * 256 + tid];
  xin[(size_t)b * 512 + 256 + tid] = s;
  xin[(size_t)b * 512 + tid] = embA[((size_t)b * 128 + t) * 256 + tid];
}

// ---------------------------------------------------------------------------
// LSTM pointwise + bn_dec (torch gate order i,f,g,o)
// ---------------------------------------------------------------------------
__global__ void lstm_pointwise(const float* __restrict__ g,
                               float* __restrict__ h, float* __restrict__ c,
                               float* __restrict__ hn,
                               const float* __restrict__ bng,
                               const float* __restrict__ bnb,
                               const float* __restrict__ bnm,
                               const float* __restrict__ bnv)
{
  const int idx = blockIdx.x * blockDim.x + threadIdx.x;
  if (idx >= 16 * 512) return;
  const int b = idx / 512, d = idx % 512;
  const float* gb = g + (size_t)b * 2048;
  const float gi = gb[d], gf = gb[512 + d], gg = gb[1024 + d], go = gb[1536 + d];
  const float cp = c[idx];
  const float c2 = sigmoidf_(gf) * cp + sigmoidf_(gi) * tanhf(gg);
  const float h2 = sigmoidf_(go) * tanhf(c2);
  c[idx] = c2;
  h[idx] = h2;
  const float scl = bng[d] * rsqrtf(bnv[d] + EPSV);
  hn[idx] = (h2 - bnm[d]) * scl + bnb[d];
}

// ===========================================================================
// Host orchestration
// ===========================================================================
struct BNP { const float *g, *b, *m, *v; };
struct Blk { const float* c1; BNP b1; const float* c2; BNP b2; const float* dc; BNP db; bool down; };

static void launch_conv(hipStream_t s, const float* x, const _Float16* wp, const BNP& bn,
                        const float* res, float* out,
                        int Bn, int IC, int IH, int IW, int OC, int OH, int OW,
                        int KH, int KW, int stride, int pad, int relu)
{
  const int blocks = ((Bn * OH * OW) / 16) * (OC / 64);
  conv_bn_wmma<<<blocks, 32, 0, s>>>(x, wp, bn.g, bn.b, bn.m, bn.v, res, out,
                                     IC, IH, IW, OC, OH, OW,
                                     KH, KW, stride, pad, relu);
}

static void launch_gemm(hipStream_t s, const float* A, int lda, const _Float16* Wh,
                        const float* bias, float* C, int ldc,
                        int M, int N, int K, int relu, int accum)
{
  const int blocks = (M / 16) * ((N + 63) / 64);
  gemm_wmma<<<blocks, 32, 0, s>>>(A, lda, Wh, bias, C, ldc, N, K, relu, accum);
}

extern "C" void kernel_launch(void* const* d_in, const int* in_sizes, int n_in,
                              void* d_out, int out_size, void* d_ws, size_t ws_size,
                              hipStream_t stream)
{
  (void)in_sizes; (void)n_in; (void)out_size; (void)ws_size;

  const float* images   = (const float*)d_in[0];   // [16,3,128,512] NCHW
  const int*   captions = (const int*)  d_in[1];   // [16,128]
  // d_in[2] = caption_lengths (unused; all T)

  // ---- params, recursive insertion order of setup_inputs()["params"] ----
  int pi = 3;
  auto F   = [&](void) -> const float* { return (const float*)d_in[pi++]; };
  auto BNF = [&](void) -> BNP { BNP r; r.g = F(); r.b = F(); r.m = F(); r.v = F(); return r; };
  auto BLK = [&](bool down) -> Blk {
    Blk k; k.c1 = F(); k.b1 = BNF(); k.c2 = F(); k.b2 = BNF();
    k.down = down; k.dc = nullptr;
    if (down) { k.dc = F(); k.db = BNF(); }
    return k;
  };

  const float* conv1w = F(); BNP bn1 = BNF();
  Blk L[8] = { BLK(false), BLK(false),   // layer1
               BLK(true),  BLK(false),   // layer2
               BLK(true),  BLK(false),   // layer3
               BLK(true),  BLK(false) }; // layer4
  const float* headw = F(); BNP headbn = BNF();
  const float* embW  = F();
  const float* attEw = F(); const float* attEb = F();
  const float* attDw = F(); const float* attDb = F();
  const float* fullw = F(); const float* fullb = F();
  const float* wih   = F(); const float* whh   = F();
  const float* bih   = F(); const float* bhh   = F();
  BNP bndec = BNF();
  const float* fcw   = F(); const float* fcb   = F();

  // ---- workspace: byte allocator, 256B aligned ----
  char* base = (char*)d_ws;
  size_t boff = 0;
  auto WSB = [&](size_t bytes) -> void* {
    void* p = base + boff;
    boff = (boff + bytes + 255) & ~(size_t)255;
    return p;
  };
  float* imgT  = (float*)WSB((size_t)16 * 3 * 128 * 512 * 4);  // NHWC image
  float* bufA  = (float*)WSB((size_t)16 * 64 * 64 * 256 * 4);  // conv1 out / temp
  float* bufB  = (float*)WSB((size_t)16 * 64 * 32 * 128 * 4);  // ping
  float* bufC  = (float*)WSB((size_t)16 * 64 * 32 * 128 * 4);  // pong
  float* bufD  = (float*)WSB((size_t)16 * 128 * 16 * 64 * 4);  // shortcut
  float* feats = (float*)WSB((size_t)16 * 64 * 256 * 4);       // [B,P,C]
  float* attE  = (float*)WSB((size_t)16 * 64 * 256 * 4);
  float* embA  = (float*)WSB((size_t)16 * 128 * 256 * 4);
  float* hbuf  = (float*)WSB(16 * 512 * 4);
  float* cbuf  = (float*)WSB(16 * 512 * 4);
  float* hnbuf = (float*)WSB(16 * 512 * 4);
  float* hproj = (float*)WSB(16 * 256 * 4);
  float* xin   = (float*)WSB(16 * 512 * 4);
  float* gbuf  = (float*)WSB(16 * 2048 * 4);

  // ---- weight prep: f16 conversion / conv repack (every call; deterministic)
  auto prep_w = [&](const float* w, int n) -> const _Float16* {
    _Float16* p = (_Float16*)WSB((size_t)n * 2);
    cvt_f16<<<(n + 255) / 256, 256, 0, stream>>>(w, p, n);
    return p;
  };
  auto prep_cw = [&](const float* w, int OC, int IC, int KHW) -> const _Float16* {
    const int n = OC * IC * KHW;
    _Float16* p = (_Float16*)WSB((size_t)n * 2);
    repack_convw<<<(n + 255) / 256, 256, 0, stream>>>(w, p, OC, IC, KHW, n);
    return p;
  };

  // ================= encoder =================
  {
    const int tot = 16 * 3 * 128 * 512;
    nchw_to_nhwc<<<(tot + 255) / 256, 256, 0, stream>>>(images, imgT, 3, 128 * 512, tot);
  }
  // conv1 7x7/2 + bn + relu: (16,128,512,3) -> (16,64,256,64)
  const _Float16* conv1w_h = prep_cw(conv1w, 64, 3, 49);
  launch_conv(stream, imgT, conv1w_h, bn1, nullptr, bufA,
              16, 3, 128, 512, 64, 64, 256, 7, 7, 2, 3, 1);
  {
    const int tot = 16 * 32 * 128 * 64;
    maxpool3x3s2_nhwc<<<(tot + 255) / 256, 256, 0, stream>>>(
        bufA, bufB, 64, 64, 256, 32, 128, tot);
  }

  int chans[5] = { 64, 64, 128, 256, 512 };
  int sh = 32, sw = 128, inC = 64;
  float* cur   = bufB;
  float* freeb = bufC;
  for (int li = 0; li < 4; ++li) {
    const int outC = chans[li + 1];
    for (int bi = 0; bi < 2; ++bi) {
      Blk& K = L[li * 2 + bi];
      const int st = (bi == 0 && li > 0) ? 2 : 1;
      const int oh = sh / st, owd = sw / st;
      const _Float16* w1 = prep_cw(K.c1, outC, inC, 9);
      const _Float16* w2 = prep_cw(K.c2, outC, outC, 9);
      launch_conv(stream, cur, w1, K.b1, nullptr, bufA,
                  16, inC, sh, sw, outC, oh, owd, 3, 3, st, 1, 1);
      const float* res = cur;
      if (K.down) {
        const _Float16* wd = prep_cw(K.dc, outC, inC, 1);
        launch_conv(stream, cur, wd, K.db, nullptr, bufD,
                    16, inC, sh, sw, outC, oh, owd, 1, 1, st, 0, 0);
        res = bufD;
      }
      launch_conv(stream, bufA, w2, K.b2, res, freeb,
                  16, outC, oh, owd, outC, oh, owd, 3, 3, 1, 1, 1);
      float* tswap = cur; cur = freeb; freeb = tswap;
      sh = oh; sw = owd; inC = outC;
    }
  }
  // head 1x1 + bn + relu -> NHWC output is exactly feats [16, 64, 256]
  const _Float16* headw_h = prep_cw(headw, 256, 512, 1);
  launch_conv(stream, cur, headw_h, headbn, nullptr, feats,
              16, 512, 4, 16, 256, 4, 16, 1, 1, 1, 0, 1);

  // ================= decoder =================
  const _Float16* attEw_h = prep_w(attEw, 256 * 256);
  const _Float16* attDw_h = prep_w(attDw, 256 * 512);
  const _Float16* wih_h   = prep_w(wih,   2048 * 512);
  const _Float16* whh_h   = prep_w(whh,   2048 * 512);
  const _Float16* fcw_h   = prep_w(fcw,   1000 * 512);

  launch_gemm(stream, feats, 256, attEw_h, attEb, attE, 256, 1024, 256, 256, 0, 0);
  {
    const int tot = 16 * 128 * 256;
    embed_gather<<<(tot + 255) / 256, 256, 0, stream>>>(captions, embW, embA);
  }
  hipMemsetAsync(hbuf, 0, (size_t)16 * 512 * sizeof(float), stream);
  hipMemsetAsync(cbuf, 0, (size_t)16 * 512 * sizeof(float), stream);

  float* out = (float*)d_out;  // [16, 128, 1000]
  for (int t = 0; t < 128; ++t) {
    launch_gemm(stream, hbuf, 512, attDw_h, attDb, hproj, 256, 16, 256, 512, 0, 0);
    attention_ctx<<<16, 256, 0, stream>>>(attE, hproj, feats, fullw, fullb, embA, t, xin);
    launch_gemm(stream, xin,  512, wih_h, bih, gbuf, 2048, 16, 2048, 512, 0, 0);
    launch_gemm(stream, hbuf, 512, whh_h, bhh, gbuf, 2048, 16, 2048, 512, 0, 1);
    lstm_pointwise<<<(16 * 512 + 255) / 256, 256, 0, stream>>>(
        gbuf, hbuf, cbuf, hnbuf, bndec.g, bndec.b, bndec.m, bndec.v);
    launch_gemm(stream, hnbuf, 512, fcw_h, fcb, out + (size_t)t * 1000, 128 * 1000,
                16, 1000, 512, 0, 0);
  }
}
